// Cyknet_58772332478694
// MI455X (gfx1250) — compile-verified, matrix operand order checked by reading, not run on hardware
//
#include <hip/hip_runtime.h>

// ---------------------------------------------------------------------------
// CYK network on MI455X (gfx1250).
//  - bf16 WMMA 16x16x32, f32 accumulate (balances ~1.5 TFLOP of GEMM against
//    L2/HBM traffic; fp32 WMMA (16x16x4) would be ~8x more instruction-bound).
//  - bf16 mirror of the CYK table in workspace: staging becomes a raw byte
//    copy -> done with gfx1250 async-to-LDS (ASYNCcnt) + double buffering.
//  - Weight B-fragments read straight from row-major [N][K] bf16 copies
//    (matches WMMA A/B per-lane register layout; no transpose staging).
// Workspace requirement: ~134 MiB bf16 table mirror + 4 MiB converted
// weights/activations.
// ---------------------------------------------------------------------------

#define LQ 64
#define BQ 32
#define DQ 512
#define FQ 512
#define K1 1024              // 2*DQ (concat K of the scores GEMM)
#define KW 264               // LDS window stride in bf16 elems (528B: 16B-aligned, bank-skewed)

typedef __attribute__((ext_vector_type(16))) __bf16 v16bf;
typedef __attribute__((ext_vector_type(8)))  float  v8f;
typedef __attribute__((ext_vector_type(4)))  float  f32x4;

union Frag { v16bf v; f32x4 q[2]; };

__device__ __forceinline__ __bf16 cvt_bf16(float f) {
  unsigned u = __builtin_bit_cast(unsigned, f);
  unsigned r = u + 0x7FFFu + ((u >> 16) & 1u);          // round-to-nearest-even
  return __builtin_bit_cast(__bf16, (unsigned short)(r >> 16));
}

// Issue async copies of one 32x256 bf16 window (global row stride 512 elems)
// into an LDS window (row stride KW). Each lane moves 16B per instruction;
// 512 threads -> 2 instructions per wave per slab (ASYNCcnt += 2).
__device__ __forceinline__ void issue_slab(const __bf16* __restrict__ src,
                                           unsigned lds_base, int tid) {
  #pragma unroll
  for (int i = 0; i < 2; ++i) {
    int c16  = tid + i * 512;          // 16B chunk id, 0..1023
    int row  = c16 >> 5;               // 32 chunks of 8 elems per 256-col row
    int col8 = (c16 & 31) * 8;
    const void* g = src + row * DQ + col8;
    unsigned    l = lds_base + (unsigned)(row * (KW * 2) + col8 * 2);
    asm volatile("global_load_async_to_lds_b128 %0, %1, off"
                 :: "v"(l), "v"(g) : "memory");
  }
}

// One wave computes a 32x32 slab (2 M-tiles x 2 N-tiles) over one 256-wide
// K window: C += A(32x256, LDS bf16) * W^T (W bf16 row-major [N][wstride],
// columns koff..koff+255).
__device__ __forceinline__ void gemm_slab(const __bf16 (*S)[KW],
                                          const __bf16* __restrict__ W,
                                          int wstride, int koff,
                                          int wave, int lane, v8f acc[2][2]) {
  const int g = lane >> 4;
  const int m = lane & 15;
  #pragma unroll 4
  for (int kk = 0; kk < 256; kk += 32) {
    Frag a0, a1;
    a0.q[0] = *(const f32x4*)&S[m][kk + 8 * g];
    a0.q[1] = *(const f32x4*)&S[m][kk + 16 + 8 * g];
    a1.q[0] = *(const f32x4*)&S[16 + m][kk + 8 * g];
    a1.q[1] = *(const f32x4*)&S[16 + m][kk + 16 + 8 * g];
    #pragma unroll
    for (int nt = 0; nt < 2; ++nt) {
      int n = wave * 32 + nt * 16 + m;
      const __bf16* bp = W + (size_t)n * wstride + koff + kk;
      Frag b;
      b.q[0] = *(const f32x4*)(bp + 8 * g);
      b.q[1] = *(const f32x4*)(bp + 16 + 8 * g);
      acc[0][nt] = __builtin_amdgcn_wmma_f32_16x16x32_bf16(
          false, a0.v, false, b.v, (short)0, acc[0][nt], false, false);
      acc[1][nt] = __builtin_amdgcn_wmma_f32_16x16x32_bf16(
          false, a1.v, false, b.v, (short)0, acc[1][nt], false, false);
    }
  }
}

// ---------------------------------------------------------------------------
// Small kernels
// ---------------------------------------------------------------------------

__global__ void __launch_bounds__(256) cyk_convert(
    const float* __restrict__ Wc, const float* __restrict__ W2,
    const float* __restrict__ W1, const float* __restrict__ xs_h,
    __bf16* __restrict__ Wc_bf, __bf16* __restrict__ W2_bf,
    __bf16* __restrict__ W1_bf, __bf16* __restrict__ xs_bf) {
  const int NC = FQ * K1, N2 = DQ * FQ, N1 = DQ * DQ, NX = LQ * BQ * DQ;
  for (int i = blockIdx.x * blockDim.x + threadIdx.x; i < NC + N2 + N1 + NX;
       i += gridDim.x * blockDim.x) {
    if (i < NC)                 Wc_bf[i]                 = cvt_bf16(Wc[i]);
    else if (i < NC + N2)       W2_bf[i - NC]            = cvt_bf16(W2[i - NC]);
    else if (i < NC + N2 + N1)  W1_bf[i - NC - N2]       = cvt_bf16(W1[i - NC - N2]);
    else                        xs_bf[i - NC - N2 - N1]  = cvt_bf16(xs_h[i - NC - N2 - N1]);
  }
}

__global__ void __launch_bounds__(256) cyk_zero_table(float* __restrict__ table) {
  const size_t n4 = (size_t)LQ * LQ * BQ * DQ / 4;
  f32x4 z = {0.f, 0.f, 0.f, 0.f};
  for (size_t i = blockIdx.x * blockDim.x + threadIdx.x; i < n4;
       i += (size_t)gridDim.x * blockDim.x)
    ((f32x4*)table)[i] = z;
}

__global__ void __launch_bounds__(256) cyk_mask(const float* __restrict__ xs_mask,
                                                float* __restrict__ mask) {
  for (int i = blockIdx.x * blockDim.x + threadIdx.x; i < LQ * LQ * BQ;
       i += gridDim.x * blockDim.x) {
    int r = i >> 11;
    int c = (i >> 5) & (LQ - 1);
    int b = i & (BQ - 1);
    float len = 0.f;
    #pragma unroll 8
    for (int l = 0; l < LQ; ++l) len += xs_mask[l * BQ + b];
    mask[i] = ((c >= r) && ((float)c < len)) ? 1.0f : 0.0f;
  }
}

// ---------------------------------------------------------------------------
// table[i,i] = xs_h[i] @ W1^T + b1   (async-staged bf16 xs, WMMA)
// ---------------------------------------------------------------------------
__global__ void __launch_bounds__(512, 1) cyk_diag(
    float* __restrict__ table, __bf16* mirror,
    const __bf16* xs_bf, const __bf16* __restrict__ W1_bf,
    const float* __restrict__ b1) {
  __shared__ __attribute__((aligned(16))) __bf16 S[2][BQ][KW];
  const int i = blockIdx.x;
  const int tid = threadIdx.x, wave = tid >> 5, lane = tid & 31;
  const int g = lane >> 4, m = lane & 15;

  const __bf16* xcell = xs_bf + (size_t)i * BQ * DQ;
  unsigned lds[2] = {(unsigned)(size_t)&S[0][0][0], (unsigned)(size_t)&S[1][0][0]};

  v8f acc[2][2];
  #pragma unroll
  for (int mt = 0; mt < 2; ++mt)
    #pragma unroll
    for (int nt = 0; nt < 2; ++nt)
      #pragma unroll
      for (int r = 0; r < 8; ++r) acc[mt][nt][r] = 0.f;

  issue_slab(xcell, lds[0], tid);
  #pragma unroll
  for (int t = 0; t < 2; ++t) {
    __syncthreads();
    if (t == 0) {
      issue_slab(xcell + 256, lds[1], tid);
      asm volatile("s_wait_asynccnt 0x2" ::: "memory");
    } else {
      asm volatile("s_wait_asynccnt 0x0" ::: "memory");
    }
    __syncthreads();
    gemm_slab(S[t], W1_bf, DQ, t * 256, wave, lane, acc);
  }

  float*  out = table  + ((size_t)(i * LQ + i) * BQ * DQ);
  __bf16* mcl = mirror + ((size_t)(i * LQ + i) * BQ * DQ);
  #pragma unroll
  for (int mt = 0; mt < 2; ++mt)
    #pragma unroll
    for (int nt = 0; nt < 2; ++nt) {
      int col = wave * 32 + nt * 16 + m;
      float bv = b1[col];
      #pragma unroll
      for (int r = 0; r < 8; ++r) {
        int row = mt * 16 + 8 * g + r;
        float v = acc[mt][nt][r] + bv;
        out[row * DQ + col] = v;
        mcl[row * DQ + col] = cvt_bf16(v);
      }
    }
}

// ---------------------------------------------------------------------------
// One workgroup per j: table[j, j+s].
// Pipeline over T = 4*s slabs (left/right cell x 2 K-windows), double-buffered
// async-to-LDS staging from the bf16 mirror.
// ---------------------------------------------------------------------------
__global__ void __launch_bounds__(512, 1) cyk_cell(
    float* table, __bf16* mirror,
    const __bf16* __restrict__ Wc_bf, const __bf16* __restrict__ W2_bf,
    const float* __restrict__ bc, const float* __restrict__ b2, int s) {
  __shared__ __attribute__((aligned(16))) __bf16 S[2][BQ][KW];
  const int j = blockIdx.x;
  const int tid = threadIdx.x, wave = tid >> 5, lane = tid & 31;
  const int g = lane >> 4, m = lane & 15;

  unsigned lds[2] = {(unsigned)(size_t)&S[0][0][0], (unsigned)(size_t)&S[1][0][0]};

  // slab t: k = t>>2, half = (t>>1)&1 (left/right cell), w = t&1 (K window)
  auto slab_src = [&](int t) -> const __bf16* {
    int k = t >> 2, half = (t >> 1) & 1, w = t & 1;
    int row = half ? (j + k + 1) : j;
    int col = half ? (j + s)     : (j + k);
    return mirror + ((size_t)(row * LQ + col) * BQ * DQ) + w * 256;
  };

  v8f mx[2][2], cacc[2][2];
  #pragma unroll
  for (int mt = 0; mt < 2; ++mt)
    #pragma unroll
    for (int nt = 0; nt < 2; ++nt)
      #pragma unroll
      for (int r = 0; r < 8; ++r) mx[mt][nt][r] = -3.0e38f;

  const int T = 4 * s;
  issue_slab(slab_src(0), lds[0], tid);

  for (int t = 0; t < T; ++t) {
    if ((t & 3) == 0) {
      #pragma unroll
      for (int mt = 0; mt < 2; ++mt)
        #pragma unroll
        for (int nt = 0; nt < 2; ++nt)
          #pragma unroll
          for (int r = 0; r < 8; ++r) cacc[mt][nt][r] = 0.f;
    }

    __syncthreads();                              // buffer[(t+1)&1] free
    if (t + 1 < T) {
      issue_slab(slab_src(t + 1), lds[(t + 1) & 1], tid);
      asm volatile("s_wait_asynccnt 0x2" ::: "memory");   // slab t landed
    } else {
      asm volatile("s_wait_asynccnt 0x0" ::: "memory");
    }
    __syncthreads();                              // slab t visible to all waves

    int koff = ((t >> 1) & 1) * DQ + (t & 1) * 256;
    gemm_slab(S[t & 1], Wc_bf, K1, koff, wave, lane, cacc);

    if ((t & 3) == 3) {
      #pragma unroll
      for (int mt = 0; mt < 2; ++mt)
        #pragma unroll
        for (int nt = 0; nt < 2; ++nt)
          #pragma unroll
          for (int r = 0; r < 8; ++r)
            mx[mt][nt][r] = fmaxf(mx[mt][nt][r], cacc[mt][nt][r]);
    }
  }

  // node = max_k(scores) + bc, packed bf16 into the two K windows:
  // cols 0..255 -> S[0], cols 256..511 -> S[1]
  __syncthreads();
  #pragma unroll
  for (int mt = 0; mt < 2; ++mt)
    #pragma unroll
    for (int nt = 0; nt < 2; ++nt) {
      int col = wave * 32 + nt * 16 + m;
      float bcv = bc[col];
      #pragma unroll
      for (int r = 0; r < 8; ++r) {
        int row = mt * 16 + 8 * g + r;
        S[col >> 8][row][col & 255] = cvt_bf16(mx[mt][nt][r] + bcv);
      }
    }
  __syncthreads();

  // vals = node @ W2^T + b2
  v8f oacc[2][2];
  #pragma unroll
  for (int mt = 0; mt < 2; ++mt)
    #pragma unroll
    for (int nt = 0; nt < 2; ++nt)
      #pragma unroll
      for (int r = 0; r < 8; ++r) oacc[mt][nt][r] = 0.f;

  gemm_slab(S[0], W2_bf, FQ, 0,   wave, lane, oacc);
  gemm_slab(S[1], W2_bf, FQ, 256, wave, lane, oacc);

  float*  out = table  + ((size_t)(j * LQ + (j + s)) * BQ * DQ);
  __bf16* mcl = mirror + ((size_t)(j * LQ + (j + s)) * BQ * DQ);
  #pragma unroll
  for (int mt = 0; mt < 2; ++mt)
    #pragma unroll
    for (int nt = 0; nt < 2; ++nt) {
      int col = wave * 32 + nt * 16 + m;
      float bv = b2[col];
      #pragma unroll
      for (int r = 0; r < 8; ++r) {
        int row = mt * 16 + 8 * g + r;
        float v = oacc[mt][nt][r] + bv;
        out[row * DQ + col] = v;
        mcl[row * DQ + col] = cvt_bf16(v);
      }
    }
}

// ---------------------------------------------------------------------------
// Host entry
// ---------------------------------------------------------------------------
extern "C" void kernel_launch(void* const* d_in, const int* in_sizes, int n_in,
                              void* d_out, int out_size, void* d_ws, size_t ws_size,
                              hipStream_t stream) {
  const float* xs_h    = (const float*)d_in[0];
  const float* xs_mask = (const float*)d_in[1];
  const float* W1      = (const float*)d_in[2];
  const float* b1      = (const float*)d_in[3];
  const float* Wc      = (const float*)d_in[4];
  const float* bc      = (const float*)d_in[5];
  const float* W2      = (const float*)d_in[6];
  const float* b2      = (const float*)d_in[7];

  float* table = (float*)d_out;
  float* mask  = table + (size_t)LQ * LQ * BQ * DQ;

  // Workspace layout (bytes):
  //   Wc_bf 1 MiB | W2_bf 512 KiB | W1_bf 512 KiB | xs_bf 2 MiB | mirror 128 MiB
  char* ws = (char*)d_ws;
  __bf16* Wc_bf  = (__bf16*)ws;
  __bf16* W2_bf  = (__bf16*)(ws + (size_t)FQ * K1 * 2);
  __bf16* W1_bf  = (__bf16*)(ws + (size_t)FQ * K1 * 2 + (size_t)DQ * FQ * 2);
  __bf16* xs_bf  = (__bf16*)(ws + (size_t)FQ * K1 * 2 + (size_t)DQ * FQ * 2 +
                             (size_t)DQ * DQ * 2);
  __bf16* mirror = (__bf16*)(ws + (size_t)FQ * K1 * 2 + (size_t)DQ * FQ * 2 +
                             (size_t)DQ * DQ * 2 + (size_t)LQ * BQ * DQ * 2);

  cyk_convert<<<2048, 256, 0, stream>>>(Wc, W2, W1, xs_h, Wc_bf, W2_bf, W1_bf, xs_bf);
  cyk_zero_table<<<2048, 256, 0, stream>>>(table);
  cyk_mask<<<512, 256, 0, stream>>>(xs_mask, mask);
  cyk_diag<<<LQ, 512, 0, stream>>>(table, mirror, xs_bf, W1_bf, b1);

  for (int s = 1; s < LQ; ++s)
    cyk_cell<<<LQ - s, 512, 0, stream>>>(table, mirror, Wc_bf, W2_bf, bc, b2, s);
}
